// FinePointMatching_35150012351354
// MI455X (gfx1250) — matrix operand order repaired and checked by reading, not compiled, not run on hardware
//
#include <hip/hip_runtime.h>
#include <hip/hip_bf16.h>

// ---------------------------------------------------------------------------
// FinePointMatching for MI455X (gfx1250, wave32, WMMA).
// All dense GEMMs lower to v_wmma_f32_16x16x32_bf16 (f32->bf16 inputs, f32 acc).
// GEMM: 8 waves (2x4), each wave owns a 32x32 output tile (2x2 WMMA tiles),
// workgroup tile 64(M) x 128(N), K-stage 64 (2 WMMA k-steps per barrier).
// LDS tiles are bf16, [m][k]/[n][k], 144-byte row stride: converts happen once
// in the global->LDS loader (v_cvt_pk_bf16_f32 + ds_store_b32), fragments are
// pure 16B-aligned ds_load_b128s.
// ---------------------------------------------------------------------------

#define B_    4
#define N_    2048
#define NF_   512
#define SLEN  513     // 1 + NF
#define NSEQ  2049    // 1 + N
#define HIDC  256
#define NH    4
#define DH    64

typedef __attribute__((ext_vector_type(8)))  float  v8f;
typedef __attribute__((ext_vector_type(16))) __bf16 v16bf;
typedef __attribute__((ext_vector_type(8)))  __bf16 v8bf;

__device__ __forceinline__ float wred_sum(float v) {
#pragma unroll
  for (int o = 16; o > 0; o >>= 1) v += __shfl_xor(v, o, 32);
  return v;
}

// Fragment = two 8-element bf16 runs of one LDS row (each 16B, 16B-aligned).
// A-operand (16x32, ISA 7.12.2): lane half h -> k = {8h..8h+7} U {16+8h..16+8h+7}
// B-operand (32x16):             lane half h -> k = {8h..8h+7} U {16+8h..16+8h+7}
__device__ __forceinline__ v16bf frag_ld(const __bf16* row, int off0, int off1) {
  v8bf lo = *(const v8bf*)(row + off0);
  v8bf hi = *(const v8bf*)(row + off1);
  v16bf f;
#pragma unroll
  for (int e = 0; e < 8; ++e) { f[e] = lo[e]; f[8 + e] = hi[e]; }
  return f;
}

__device__ __forceinline__ void store_tile(
    v8f acc, int gm0, int gn, int half, int M, int N,
    const float* bias, const float* Db, float* Cb,
    int ldc, int ldd, int relu, float alpha)
{
  if (gn >= N) return;
  float bv = bias ? bias[gn] : 0.f;
#pragma unroll
  for (int r = 0; r < 8; ++r) {
    int gm = gm0 + r + 8 * half;      // C/D: vgpr r -> M = r + 8*half
    if (gm < M) {
      float v = acc[r] * alpha + bv;
      if (Db) v += Db[(long)gm * ldd + gn];
      if (relu) v = fmaxf(v, 0.f);
      Cb[(long)gm * ldc + gn] = v;
    }
  }
}

// ---------------------------------------------------------------------------
// Generic WMMA GEMM:  C = alpha * op(A) @ op(B) + bias + Add  (optional ReLU)
// Batched via grid.z:  off = (z/zdiv)*s?o + (z%zdiv)*s?i
// ---------------------------------------------------------------------------
__global__ __launch_bounds__(256) void k_gemm(
    const float* __restrict__ A, const float* __restrict__ Bm,
    const float* __restrict__ bias, const float* __restrict__ Add,
    float* __restrict__ C,
    int M, int N, int K, int lda, int ldb, int ldc, int ldd,
    long sAo, long sAi, long sBo, long sBi, long sCo, long sCi, long sDo, long sDi,
    int zdiv, int tA, int tB, int relu, float alpha)
{
  __shared__ __bf16 As[64][72];    // [m][k], 144B row stride (16B aligned)
  __shared__ __bf16 Bs[128][72];   // [n][k]
  int z  = blockIdx.z, zo = z / zdiv, zi = z - zo * zdiv;
  const float* Ab = A  + zo * sAo + zi * sAi;
  const float* Bb = Bm + zo * sBo + zi * sBi;
  float*       Cb = C  + zo * sCo + zi * sCi;
  const float* Db = Add ? (Add + zo * sDo + zi * sDi) : (const float*)0;

  int tid = threadIdx.x, lane = tid & 31, wave = tid >> 5;
  int wm = wave >> 2, wn = wave & 3;      // 2 x 4 wave grid
  int half = lane >> 4, l16 = lane & 15;
  int m0 = blockIdx.y * 64, n0 = blockIdx.x * 128;

  v8f acc00 = {}, acc01 = {}, acc10 = {}, acc11 = {};
  for (int k0 = 0; k0 < K; k0 += 64) {
    // ---- A tile 64x64 -> As[m][k] (bf16, packed-pair stores) ----
    if (!tA) {
      for (int i = tid; i < 64 * 32; i += 256) {   // pairs along k
        int pr = i & 31, mm = i >> 5;
        int gm = m0 + mm, gk = k0 + pr * 2;
        float v0 = 0.f, v1 = 0.f;
        if (gm < M) {
          if (gk     < K) v0 = Ab[(long)gm * lda + gk];
          if (gk + 1 < K) v1 = Ab[(long)gm * lda + gk + 1];
        }
        union { __bf16 h[2]; unsigned u; } pk;
        pk.h[0] = (__bf16)v0; pk.h[1] = (__bf16)v1;
        *(unsigned*)&As[mm][pr * 2] = pk.u;
      }
    } else {
      for (int i = tid; i < 64 * 64; i += 256) {   // gm fastest (coalesced)
        int mm = i & 63, kk = i >> 6;
        int gm = m0 + mm, gk = k0 + kk;
        float v = 0.f;
        if (gm < M && gk < K) v = Ab[(long)gk * lda + gm];
        As[mm][kk] = (__bf16)v;
      }
    }
    // ---- B tile 64x128 -> Bs[n][k] ----
    if (!tB) {
      for (int i = tid; i < 128 * 32; i += 256) {  // gn fastest, pairs along k
        int nn = i & 127, pr = i >> 7;
        int gn = n0 + nn, gk = k0 + pr * 2;
        float v0 = 0.f, v1 = 0.f;
        if (gn < N) {
          if (gk     < K) v0 = Bb[(long)gk * ldb + gn];
          if (gk + 1 < K) v1 = Bb[(long)(gk + 1) * ldb + gn];
        }
        union { __bf16 h[2]; unsigned u; } pk;
        pk.h[0] = (__bf16)v0; pk.h[1] = (__bf16)v1;
        *(unsigned*)&Bs[nn][pr * 2] = pk.u;
      }
    } else {
      for (int i = tid; i < 128 * 64; i += 256) {  // gk fastest (coalesced)
        int nn = i >> 6, kk = i & 63;
        int gn = n0 + nn, gk = k0 + kk;
        float v = 0.f;
        if (gn < N && gk < K) v = Bb[(long)gn * ldb + gk];
        Bs[nn][kk] = (__bf16)v;
      }
    }
    int nk = k0 + 64;
    if (nk < K) {  // gfx1250 global_prefetch_b8 of next K-stage (speculative-safe)
      int pm = m0 + (tid & 63); if (pm >= M) pm = M - 1;
      int pn = n0 + (tid & 127); if (pn >= N) pn = N - 1;
      __builtin_prefetch(tA ? &Ab[(long)nk * lda + pm] : &Ab[(long)pm * lda + nk], 0, 1);
      __builtin_prefetch(tB ? &Bb[(long)pn * ldb + nk] : &Bb[(long)nk * ldb + pn], 0, 1);
    }
    __syncthreads();
    // ---- 2 WMMA k-steps per stage; fragments are pure ds_load_b128 ----
    const __bf16* ar0 = &As[wm * 32 + l16][0];
    const __bf16* ar1 = &As[wm * 32 + 16 + l16][0];
    const __bf16* br0 = &Bs[wn * 32 + l16][0];
    const __bf16* br1 = &Bs[wn * 32 + 16 + l16][0];
#pragma unroll
    for (int ks = 0; ks < 64; ks += 32) {
      v16bf a0 = frag_ld(ar0, ks + 8 * half, ks + 16 + 8 * half);
      v16bf a1 = frag_ld(ar1, ks + 8 * half, ks + 16 + 8 * half);
      v16bf b0 = frag_ld(br0, ks + 8 * half, ks + 16 + 8 * half);
      v16bf b1 = frag_ld(br1, ks + 8 * half, ks + 16 + 8 * half);
      acc00 = __builtin_amdgcn_wmma_f32_16x16x32_bf16(false, a0, false, b0, (short)0, acc00, false, false);
      acc01 = __builtin_amdgcn_wmma_f32_16x16x32_bf16(false, a0, false, b1, (short)0, acc01, false, false);
      acc10 = __builtin_amdgcn_wmma_f32_16x16x32_bf16(false, a1, false, b0, (short)0, acc10, false, false);
      acc11 = __builtin_amdgcn_wmma_f32_16x16x32_bf16(false, a1, false, b1, (short)0, acc11, false, false);
    }
    __syncthreads();
  }
  int gma = m0 + wm * 32, gmb = gma + 16;
  int gna = n0 + wn * 32 + l16, gnb = gna + 16;
  store_tile(acc00, gma, gna, half, M, N, bias, Db, Cb, ldc, ldd, relu, alpha);
  store_tile(acc01, gma, gnb, half, M, N, bias, Db, Cb, ldc, ldd, relu, alpha);
  store_tile(acc10, gmb, gna, half, M, N, bias, Db, Cb, ldc, ldd, relu, alpha);
  store_tile(acc11, gmb, gnb, half, M, N, bias, Db, Cb, ldc, ldd, relu, alpha);
}

// ---------------------------------------------------------------------------
// p1_ = (p1 - init_t) @ init_R
// ---------------------------------------------------------------------------
__global__ void k_xform(const float* __restrict__ p, const float* __restrict__ Rm,
                        const float* __restrict__ t, float* __restrict__ o)
{
  int i = blockIdx.x * blockDim.x + threadIdx.x;
  if (i >= B_ * N_) return;
  int b = i / N_;
  float x = p[i * 3 + 0] - t[b * 3 + 0];
  float y = p[i * 3 + 1] - t[b * 3 + 1];
  float z = p[i * 3 + 2] - t[b * 3 + 2];
  const float* R = Rm + b * 9;
  o[i * 3 + 0] = x * R[0] + y * R[3] + z * R[6];
  o[i * 3 + 1] = x * R[1] + y * R[4] + z * R[7];
  o[i * 3 + 2] = x * R[2] + y * R[5] + z * R[8];
}

// ---------------------------------------------------------------------------
// Fused ball-query + shared MLP (6->32->64->128, ReLU) + max over samples.
// One thread per query point; neighbor scan is wave-uniform (scalar loads).
// ---------------------------------------------------------------------------
__device__ __forceinline__ void pe_sample(
    float i0, float i1, float i2, float i3, float i4, float i5,
    const float* sW1, const float* sB1, const float* sW2, const float* sB2,
    const float* sW3, const float* sB3, float* mx)
{
  float in6[6] = {i0, i1, i2, i3, i4, i5};
  float h1[32];
  for (int o = 0; o < 32; ++o) {
    float a = sB1[o];
    for (int c = 0; c < 6; ++c) a += sW1[o * 6 + c] * in6[c];
    h1[o] = fmaxf(a, 0.f);
  }
  float h2[64];
  for (int o = 0; o < 64; ++o) {
    float a = sB2[o];
    for (int c = 0; c < 32; ++c) a += sW2[o * 32 + c] * h1[c];
    h2[o] = fmaxf(a, 0.f);
  }
  for (int o = 0; o < 128; ++o) {
    float a = sB3[o];
    for (int c = 0; c < 64; ++c) a += sW3[o * 64 + c] * h2[c];
    a = fmaxf(a, 0.f);
    mx[o] = fmaxf(mx[o], a);
  }
}

__global__ __launch_bounds__(256) void k_pe_mlp(
    const float* __restrict__ pts,
    const float* __restrict__ W1, const float* __restrict__ b1,
    const float* __restrict__ W2, const float* __restrict__ b2,
    const float* __restrict__ W3, const float* __restrict__ b3,
    float rad, int nsample, float* __restrict__ outf)
{
  __shared__ float sW1[32 * 6], sB1[32], sW2[64 * 32], sB2[64], sW3[128 * 64], sB3[128];
  int tid = threadIdx.x;
  for (int i = tid; i < 192;  i += 256) sW1[i] = W1[i];
  for (int i = tid; i < 32;   i += 256) sB1[i] = b1[i];
  for (int i = tid; i < 2048; i += 256) sW2[i] = W2[i];
  for (int i = tid; i < 64;   i += 256) sB2[i] = b2[i];
  for (int i = tid; i < 8192; i += 256) sW3[i] = W3[i];
  for (int i = tid; i < 128;  i += 256) sB3[i] = b3[i];
  __syncthreads();

  int b = blockIdx.y;
  int n = blockIdx.x * 256 + tid;
  if (n >= N_) return;
  const float* pb = pts + (long)b * N_ * 3;
  float qx = pb[n * 3 + 0], qy = pb[n * 3 + 1], qz = pb[n * 3 + 2];
  float r2 = rad * rad;
  float mx[128];
#pragma unroll
  for (int o = 0; o < 128; ++o) mx[o] = -1e30f;
  int hits = 0;
  for (int j = 0; j < N_; ++j) {       // first `nsample` in-radius by index order
    float px = pb[j * 3 + 0], py = pb[j * 3 + 1], pz = pb[j * 3 + 2];
    float dx = px - qx, dy = py - qy, dz = pz - qz;
    float d2 = dx * dx + dy * dy + dz * dz;
    if (d2 < r2 && hits < nsample) {
      ++hits;
      pe_sample(dx, dy, dz, px, py, pz, sW1, sB1, sW2, sB2, sW3, sB3, mx);
    }
  }
  if (hits == 0) {                     // reference fallback: index 0
    float px = pb[0], py = pb[1], pz = pb[2];
    pe_sample(px - qx, py - qy, pz - qz, px, py, pz, sW1, sB1, sW2, sB2, sW3, sB3, mx);
  }
  float* op = outf + ((long)b * N_ + n) * HIDC;
  for (int o = 0; o < 128; ++o) op[o] = mx[o];
}

// ---------------------------------------------------------------------------
// Small structural kernels
// ---------------------------------------------------------------------------
__global__ void k_bgrow(float* __restrict__ x, const float* __restrict__ bg)
{
  x[(long)blockIdx.x * NSEQ * HIDC + threadIdx.x] = bg[threadIdx.x];
}

__global__ __launch_bounds__(256) void k_gather(
    const float* __restrict__ x, const float* __restrict__ geo,
    const int* __restrict__ fps, float* __restrict__ s)
{
  int c = threadIdx.x, i = blockIdx.x, b = blockIdx.y;
  float v;
  if (i == 0) v = x[(long)b * NSEQ * HIDC + c];
  else {
    int j = fps[b * NF_ + (i - 1)];
    v = x[((long)b * NSEQ + 1 + j) * HIDC + c] + geo[((long)b * N_ + j) * HIDC + c];
  }
  s[((long)b * SLEN + i) * HIDC + c] = v;
}

__global__ void k_copyrow0(float* __restrict__ x, const float* __restrict__ s)
{
  x[(long)blockIdx.x * NSEQ * HIDC + threadIdx.x] =
      s[(long)blockIdx.x * SLEN * HIDC + threadIdx.x];
}

// ---------------------------------------------------------------------------
// Focusing: relu+eps, keep norm, cube, renorm. One wave per (row, head).
// ---------------------------------------------------------------------------
__global__ __launch_bounds__(256) void k_focus(float* __restrict__ X, int nrows)
{
  int gid = blockIdx.x * 256 + threadIdx.x;
  int gw = gid >> 5, lane = gid & 31;
  if (gw >= nrows * NH) return;
  long base = (long)(gw >> 2) * HIDC + (gw & 3) * DH;
  float a = fmaxf(X[base + lane],      0.f) + 1e-6f;
  float c = fmaxf(X[base + lane + 32], 0.f) + 1e-6f;
  float n1 = sqrtf(wred_sum(a * a + c * c));
  float pa = a * a * a, pc = c * c * c;     // FOCUS = 3
  float n2 = sqrtf(wred_sum(pa * pa + pc * pc));
  float sc = n1 / (n2 + 1e-6f);
  X[base + lane]      = pa * sc;
  X[base + lane + 32] = pc * sc;
}

__global__ __launch_bounds__(256) void k_ksum(const float* __restrict__ Km, int Nk,
                                              float* __restrict__ ks)
{
  __shared__ float part[4][64];
  int b = blockIdx.x, h = blockIdx.y;
  int t = threadIdx.x, d = t & 63, seg = t >> 6;
  float s = 0.f;
  for (int n = seg; n < Nk; n += 4)
    s += Km[((long)b * Nk + n) * HIDC + h * DH + d];
  part[seg][d] = s;
  __syncthreads();
  if (seg == 0)
    ks[(b * NH + h) * DH + d] = part[0][d] + part[1][d] + part[2][d] + part[3][d];
}

// out = (q @ kv) * 1/(q . ksum + eps), one wave per (row, head)
__global__ __launch_bounds__(256) void k_linattn(
    const float* __restrict__ Q, const float* __restrict__ kv,
    const float* __restrict__ ks, float* __restrict__ AO, int Nq)
{
  int gid = blockIdx.x * 256 + threadIdx.x;
  int gw = gid >> 5, lane = gid & 31;
  if (gw >= B_ * Nq * NH) return;
  int row = gw >> 2, h = gw & 3;
  int b = row / Nq;
  long qb  = (long)row * HIDC + h * DH;
  long kvb = ((long)b * NH + h) * DH * DH;
  int  ksb = (b * NH + h) * DH;
  float q0 = Q[qb + lane], q1 = Q[qb + lane + 32];
  float den = wred_sum(q0 * ks[ksb + lane] + q1 * ks[ksb + lane + 32]);
  float z = 1.f / (den + 1e-6f);
  float a0 = 0.f, a1 = 0.f;
  for (int d = 0; d < DH; ++d) {
    float qd = Q[qb + d];                 // wave-uniform (scalarizes)
    a0 += qd * kv[kvb + d * DH + lane];
    a1 += qd * kv[kvb + d * DH + lane + 32];
  }
  AO[qb + lane]      = a0 * z;
  AO[qb + lane + 32] = a1 * z;
}

__global__ __launch_bounds__(256) void k_lnorm(
    const float* __restrict__ X, float* __restrict__ Yd,
    const float* __restrict__ g, const float* __restrict__ bb, int nrows)
{
  int gid = blockIdx.x * 256 + threadIdx.x;
  int gw = gid >> 5, lane = gid & 31;
  if (gw >= nrows) return;
  long base = (long)gw * HIDC;
  float v[8]; float s = 0.f;
#pragma unroll
  for (int i = 0; i < 8; ++i) { v[i] = X[base + lane + 32 * i]; s += v[i]; }
  float mu = wred_sum(s) * (1.f / HIDC);
  float vs = 0.f;
#pragma unroll
  for (int i = 0; i < 8; ++i) { float d = v[i] - mu; vs += d * d; }
  float var = wred_sum(vs) * (1.f / HIDC);
  float rstd = rsqrtf(var + 1e-5f);
#pragma unroll
  for (int i = 0; i < 8; ++i) {
    int c = lane + 32 * i;
    Yd[base + c] = (v[i] - mu) * rstd * g[c] + bb[c];
  }
}

__global__ __launch_bounds__(256) void k_rownorm(float* __restrict__ X, int nrows)
{
  int gid = blockIdx.x * 256 + threadIdx.x;
  int gw = gid >> 5, lane = gid & 31;
  if (gw >= nrows) return;
  long base = (long)gw * HIDC;
  float v[8]; float ss = 0.f;
#pragma unroll
  for (int i = 0; i < 8; ++i) { v[i] = X[base + lane + 32 * i]; ss += v[i] * v[i]; }
  float inv = 1.f / (sqrtf(wred_sum(ss)) + 1e-6f);
#pragma unroll
  for (int i = 0; i < 8; ++i) X[base + lane + 32 * i] = v[i] * inv;
}

// rows 1..N_ of the (NSEQ x NSEQ) attention matrix, softmax over all NSEQ cols
__global__ __launch_bounds__(256) void k_softmax(float* __restrict__ at)
{
  __shared__ float red[256];
  int b = blockIdx.y, n = blockIdx.x + 1;
  long base = ((long)b * NSEQ + n) * NSEQ;
  int tid = threadIdx.x;
  float mx = -1e30f;
  for (int c = tid; c < NSEQ; c += 256) mx = fmaxf(mx, at[base + c]);
  red[tid] = mx; __syncthreads();
  for (int s = 128; s > 0; s >>= 1) { if (tid < s) red[tid] = fmaxf(red[tid], red[tid + s]); __syncthreads(); }
  mx = red[0]; __syncthreads();
  float sum = 0.f;
  for (int c = tid; c < NSEQ; c += 256) {
    float e = __expf(at[base + c] - mx);
    at[base + c] = e; sum += e;
  }
  red[tid] = sum; __syncthreads();
  for (int s = 128; s > 0; s >>= 1) { if (tid < s) red[tid] += red[tid + s]; __syncthreads(); }
  float inv = 1.f / red[0];
  for (int c = tid; c < NSEQ; c += 256) at[base + c] *= inv;
}

// p2_hat, weights. One wave per (b, n).
__global__ __launch_bounds__(256) void k_p2hat(
    const float* __restrict__ at, const float* __restrict__ p2,
    float* __restrict__ p2h, float* __restrict__ wb)
{
  int gid = blockIdx.x * 256 + threadIdx.x;
  int gw = gid >> 5, lane = gid & 31;
  if (gw >= B_ * N_) return;
  int b = gw / N_, n = gw % N_;
  long base = ((long)b * NSEQ + n + 1) * NSEQ;
  float s = 0.f, sx = 0.f, sy = 0.f, sz = 0.f;
  for (int m = lane; m < N_; m += 32) {
    float pr = at[base + 1 + m];
    const float* pp = p2 + ((long)b * N_ + m) * 3;
    s += pr; sx += pr * pp[0]; sy += pr * pp[1]; sz += pr * pp[2];
  }
  s = wred_sum(s); sx = wred_sum(sx); sy = wred_sum(sy); sz = wred_sum(sz);
  if (lane == 0) {
    float inv = 1.f / (s + 1e-6f);
    long o = ((long)b * N_ + n) * 3;
    p2h[o + 0] = sx * inv; p2h[o + 1] = sy * inv; p2h[o + 2] = sz * inv;
    wb[b * N_ + n] = 1.f - at[base + 0];
  }
}

__global__ __launch_bounds__(256) void k_stats1(
    const float* __restrict__ wbuf, const float* __restrict__ p1,
    const float* __restrict__ p2h, float* __restrict__ stats)
{
  __shared__ float red[256];
  int b = blockIdx.x, tid = threadIdx.x;
  float a[7] = {0, 0, 0, 0, 0, 0, 0};
  for (int n = tid; n < N_; n += 256) {
    float ww = wbuf[b * N_ + n];
    const float* pa = p1  + ((long)b * N_ + n) * 3;
    const float* pc = p2h + ((long)b * N_ + n) * 3;
    a[0] += ww;
    a[1] += ww * pa[0]; a[2] += ww * pa[1]; a[3] += ww * pa[2];
    a[4] += ww * pc[0]; a[5] += ww * pc[1]; a[6] += ww * pc[2];
  }
  for (int k = 0; k < 7; ++k) {
    red[tid] = a[k]; __syncthreads();
    for (int s = 128; s > 0; s >>= 1) { if (tid < s) red[tid] += red[tid + s]; __syncthreads(); }
    if (tid == 0) stats[b * 8 + k] = red[0];
    __syncthreads();
  }
}

__global__ __launch_bounds__(256) void k_stats2(
    const float* __restrict__ wbuf, const float* __restrict__ p1,
    const float* __restrict__ p2h, const float* __restrict__ stats,
    float* __restrict__ Hm)
{
  __shared__ float red[256];
  int b = blockIdx.x, tid = threadIdx.x;
  float S = stats[b * 8], inv = 1.f / (S + 1e-6f);
  float c1x = stats[b * 8 + 1] * inv, c1y = stats[b * 8 + 2] * inv, c1z = stats[b * 8 + 3] * inv;
  float c2x = stats[b * 8 + 4] * inv, c2y = stats[b * 8 + 5] * inv, c2z = stats[b * 8 + 6] * inv;
  float a[9] = {0, 0, 0, 0, 0, 0, 0, 0, 0};
  for (int n = tid; n < N_; n += 256) {
    float ww = wbuf[b * N_ + n];
    const float* pa = p1  + ((long)b * N_ + n) * 3;
    const float* pc = p2h + ((long)b * N_ + n) * 3;
    float q1x = pa[0] - c1x, q1y = pa[1] - c1y, q1z = pa[2] - c1z;
    float q2x = pc[0] - c2x, q2y = pc[1] - c2y, q2z = pc[2] - c2z;
    a[0] += ww * q1x * q2x; a[1] += ww * q1x * q2y; a[2] += ww * q1x * q2z;
    a[3] += ww * q1y * q2x; a[4] += ww * q1y * q2y; a[5] += ww * q1y * q2z;
    a[6] += ww * q1z * q2x; a[7] += ww * q1z * q2y; a[8] += ww * q1z * q2z;
  }
  for (int k = 0; k < 9; ++k) {
    red[tid] = a[k]; __syncthreads();
    for (int s = 128; s > 0; s >>= 1) { if (tid < s) red[tid] += red[tid + s]; __syncthreads(); }
    if (tid == 0) Hm[b * 9 + k] = red[0] * inv;
    __syncthreads();
  }
}

// Kabsch via Jacobi eigen of H^T H; one thread per batch.
__global__ void k_svd(const float* __restrict__ Hm, const float* __restrict__ stats,
                      const float* __restrict__ radius, float* __restrict__ rtb,
                      float* __restrict__ out)
{
  int b = threadIdx.x;
  if (b >= B_) return;
  float Hl[3][3];
  for (int i = 0; i < 3; ++i) for (int j = 0; j < 3; ++j) Hl[i][j] = Hm[b * 9 + i * 3 + j];
  float A[3][3];
  for (int i = 0; i < 3; ++i) for (int j = 0; j < 3; ++j) {
    float s = 0.f;
    for (int k = 0; k < 3; ++k) s += Hl[k][i] * Hl[k][j];
    A[i][j] = s;
  }
  float V[3][3] = {{1, 0, 0}, {0, 1, 0}, {0, 0, 1}};
  for (int sweep = 0; sweep < 16; ++sweep) {
    for (int pi = 0; pi < 3; ++pi) {
      int p = (pi == 2) ? 1 : 0;
      int q = (pi == 0) ? 1 : 2;
      float apq = A[p][q];
      if (fabsf(apq) < 1e-20f) continue;
      float th = 0.5f * atan2f(2.f * apq, A[q][q] - A[p][p]);
      float c = cosf(th), s = sinf(th);
      for (int k = 0; k < 3; ++k) { float kp = A[k][p], kq = A[k][q]; A[k][p] = c * kp - s * kq; A[k][q] = s * kp + c * kq; }
      for (int k = 0; k < 3; ++k) { float pk = A[p][k], qk = A[q][k]; A[p][k] = c * pk - s * qk; A[q][k] = s * pk + c * qk; }
      for (int k = 0; k < 3; ++k) { float vp = V[k][p], vq = V[k][q]; V[k][p] = c * vp - s * vq; V[k][q] = s * vp + c * vq; }
    }
  }
  float lam[3] = {A[0][0], A[1][1], A[2][2]};
  for (int i = 0; i < 2; ++i) for (int j = 0; j < 2 - i; ++j) {
    if (lam[j] < lam[j + 1]) {
      float tl = lam[j]; lam[j] = lam[j + 1]; lam[j + 1] = tl;
      for (int k = 0; k < 3; ++k) { float tv = V[k][j]; V[k][j] = V[k][j + 1]; V[k][j + 1] = tv; }
    }
  }
  float u0[3], u1[3];
  for (int i = 0; i < 3; ++i) u0[i] = Hl[i][0] * V[0][0] + Hl[i][1] * V[1][0] + Hl[i][2] * V[2][0];
  float n0 = sqrtf(u0[0] * u0[0] + u0[1] * u0[1] + u0[2] * u0[2]);
  if (n0 > 1e-12f) { u0[0] /= n0; u0[1] /= n0; u0[2] /= n0; }
  else { u0[0] = 1.f; u0[1] = 0.f; u0[2] = 0.f; }
  for (int i = 0; i < 3; ++i) u1[i] = Hl[i][0] * V[0][1] + Hl[i][1] * V[1][1] + Hl[i][2] * V[2][1];
  float d01 = u1[0] * u0[0] + u1[1] * u0[1] + u1[2] * u0[2];
  for (int i = 0; i < 3; ++i) u1[i] -= d01 * u0[i];
  float n1 = sqrtf(u1[0] * u1[0] + u1[1] * u1[1] + u1[2] * u1[2]);
  if (n1 > 1e-12f) { u1[0] /= n1; u1[1] /= n1; u1[2] /= n1; }
  else {
    float ax = fabsf(u0[0]) < 0.9f ? 1.f : 0.f;
    float t0[3] = {ax, 1.f - ax, 0.f};
    float dt = t0[0] * u0[0] + t0[1] * u0[1] + t0[2] * u0[2];
    for (int i = 0; i < 3; ++i) u1[i] = t0[i] - dt * u0[i];
    float nn = sqrtf(u1[0] * u1[0] + u1[1] * u1[1] + u1[2] * u1[2]);
    u1[0] /= nn; u1[1] /= nn; u1[2] /= nn;
  }
  float u2[3] = { u0[1] * u1[2] - u0[2] * u1[1],
                  u0[2] * u1[0] - u0[0] * u1[2],
                  u0[0] * u1[1] - u0[1] * u1[0] };
  float U[3][3];
  for (int i = 0; i < 3; ++i) { U[i][0] = u0[i]; U[i][1] = u1[i]; U[i][2] = u2[i]; }
  float detV = V[0][0] * (V[1][1] * V[2][2] - V[1][2] * V[2][1])
             - V[0][1] * (V[1][0] * V[2][2] - V[1][2] * V[2][0])
             + V[0][2] * (V[1][0] * V[2][1] - V[1][1] * V[2][0]);
  float D2 = detV;  // det(U) == +1 by construction
  float R[3][3];
  for (int i = 0; i < 3; ++i) for (int k = 0; k < 3; ++k)
    R[i][k] = V[i][0] * U[k][0] + V[i][1] * U[k][1] + D2 * V[i][2] * U[k][2];
  float S = stats[b * 8], inv = 1.f / (S + 1e-6f);
  float c1[3] = {stats[b * 8 + 1] * inv, stats[b * 8 + 2] * inv, stats[b * 8 + 3] * inv};
  float c2[3] = {stats[b * 8 + 4] * inv, stats[b * 8 + 5] * inv, stats[b * 8 + 6] * inv};
  float t[3];
  for (int i = 0; i < 3; ++i)
    t[i] = c2[i] - (R[i][0] * c1[0] + R[i][1] * c1[1] + R[i][2] * c1[2]);
  for (int i = 0; i < 3; ++i) for (int j = 0; j < 3; ++j) {
    out[b * 9 + i * 3 + j] = R[i][j];
    rtb[b * 12 + i * 3 + j] = R[i][j];
  }
  float rs = radius[b] + 1e-6f;
  for (int i = 0; i < 3; ++i) { out[36 + b * 3 + i] = t[i] * rs; rtb[b * 12 + 9 + i] = t[i]; }
}

__global__ __launch_bounds__(256) void k_score(
    const float* __restrict__ wbuf, const float* __restrict__ p1,
    const float* __restrict__ p2h, const float* __restrict__ stats,
    const float* __restrict__ rtb, float* __restrict__ out)
{
  __shared__ float red[256];
  int b = blockIdx.x, tid = threadIdx.x;
  float R0 = rtb[b * 12 + 0], R1 = rtb[b * 12 + 1], R2 = rtb[b * 12 + 2];
  float R3 = rtb[b * 12 + 3], R4 = rtb[b * 12 + 4], R5 = rtb[b * 12 + 5];
  float R6 = rtb[b * 12 + 6], R7 = rtb[b * 12 + 7], R8 = rtb[b * 12 + 8];
  float t0 = rtb[b * 12 + 9], t1 = rtb[b * 12 + 10], t2 = rtb[b * 12 + 11];
  float inv = 1.f / (stats[b * 8] + 1e-6f);
  float acc = 0.f;
  for (int n = tid; n < N_; n += 256) {
    const float* pa = p1  + ((long)b * N_ + n) * 3;
    const float* pc = p2h + ((long)b * N_ + n) * 3;
    float px = R0 * pa[0] + R1 * pa[1] + R2 * pa[2] + t0;
    float py = R3 * pa[0] + R4 * pa[1] + R5 * pa[2] + t1;
    float pz = R6 * pa[0] + R7 * pa[1] + R8 * pa[2] + t2;
    float dx = px - pc[0], dy = py - pc[1], dz = pz - pc[2];
    float res = sqrtf(dx * dx + dy * dy + dz * dz);
    acc += wbuf[b * N_ + n] * inv * __expf(-res);
  }
  red[tid] = acc; __syncthreads();
  for (int s = 128; s > 0; s >>= 1) { if (tid < s) red[tid] += red[tid + s]; __syncthreads(); }
  if (tid == 0) out[48 + b] = red[0];
}

// ---------------------------------------------------------------------------
// Host-side helpers
// ---------------------------------------------------------------------------
static inline void gemm(hipStream_t st, const float* A, const float* Bm, const float* bias,
                        const float* Add, float* C, int M, int N, int K,
                        int lda, int ldb, int ldc, int ldd,
                        long sAo, long sAi, long sBo, long sBi,
                        long sCo, long sCi, long sDo, long sDi,
                        int nz, int zdiv, int tA, int tB, int relu, float alpha)
{
  dim3 g((N + 127) / 128, (M + 63) / 64, nz), blk(256, 1, 1);
  k_gemm<<<g, blk, 0, st>>>(A, Bm, bias, Add, C, M, N, K, lda, ldb, ldc, ldd,
                            sAo, sAi, sBo, sBi, sCo, sCi, sDo, sDi,
                            zdiv, tA, tB, relu, alpha);
}

static inline void gemm_flat(hipStream_t st, const float* A, const float* Bm,
                             const float* bias, const float* Add, float* C,
                             int M, int N, int K, int tB, int relu, float alpha)
{
  gemm(st, A, Bm, bias, Add, C, M, N, K,
       K, tB ? K : N, N, N,
       0, 0, 0, 0, 0, 0, 0, 0, 1, 1, 0, tB, relu, alpha);
}

struct WSB { float *Q, *K, *V, *AO, *Y, *FH, *kv, *ks; };

// Linear-attention transformer block (post-LN as in reference).
static void attn_run(hipStream_t st, void* const* d_in, int pb,
                     float* dst, const float* xq, int Nq, const float* xkv, int Nk,
                     const WSB& w)
{
  auto P = [&](int i) { return (const float*)d_in[i]; };
  int Mq = B_ * Nq, Mk = B_ * Nk;
  gemm_flat(st, xq,  P(pb + 0), P(pb + 1), nullptr, w.Q, Mq, HIDC, HIDC, 0, 0, 1.f);
  gemm_flat(st, xkv, P(pb + 2), P(pb + 3), nullptr, w.K, Mk, HIDC, HIDC, 0, 0, 1.f);
  gemm_flat(st, xkv, P(pb + 4), P(pb + 5), nullptr, w.V, Mk, HIDC, HIDC, 0, 0, 1.f);
  k_focus<<<(Mq * NH + 7) / 8, 256, 0, st>>>(w.Q, Mq);
  k_focus<<<(Mk * NH + 7) / 8, 256, 0, st>>>(w.K, Mk);
  k_ksum<<<dim3(B_, NH), 256, 0, st>>>(w.K, Nk, w.ks);
  // kv[b,h] = K^T V  (64x64, contraction over sequence) via transA WMMA GEMM
  gemm(st, w.K, w.V, nullptr, nullptr, w.kv, DH, DH, Nk,
       HIDC, HIDC, DH, 0,
       (long)Nk * HIDC, DH, (long)Nk * HIDC, DH,
       (long)NH * DH * DH, (long)DH * DH, 0, 0,
       B_ * NH, NH, 1, 0, 0, 1.f);
  k_linattn<<<(Mq * NH + 7) / 8, 256, 0, st>>>(w.Q, w.kv, w.ks, w.AO, Nq);
  gemm_flat(st, w.AO, P(pb + 6), P(pb + 7), xq, w.Y, Mq, HIDC, HIDC, 0, 0, 1.f);  // + residual
  k_lnorm<<<(Mq + 7) / 8, 256, 0, st>>>(w.Y, w.Y, P(pb + 12), P(pb + 13), Mq);
  gemm_flat(st, w.Y,  P(pb + 8),  P(pb + 9),  nullptr, w.FH, Mq, 4 * HIDC, HIDC, 0, 1, 1.f);
  gemm_flat(st, w.FH, P(pb + 10), P(pb + 11), w.Y,     w.Y,  Mq, HIDC, 4 * HIDC, 0, 0, 1.f);
  k_lnorm<<<(Mq + 7) / 8, 256, 0, st>>>(w.Y, dst, P(pb + 14), P(pb + 15), Mq);
}

// ---------------------------------------------------------------------------
extern "C" void kernel_launch(void* const* d_in, const int* in_sizes, int n_in,
                              void* d_out, int out_size, void* d_ws, size_t ws_size,
                              hipStream_t stream)
{
  (void)in_sizes; (void)n_in; (void)out_size; (void)ws_size;
  const float* p1    = (const float*)d_in[0];
  const float* f1    = (const float*)d_in[1];
  const float* geo1  = (const float*)d_in[2];
  const int*   fps1  = (const int*)  d_in[3];
  const float* p2    = (const float*)d_in[4];
  const float* f2    = (const float*)d_in[5];
  const float* geo2  = (const float*)d_in[6];
  const int*   fps2  = (const int*)  d_in[7];
  const float* radius= (const float*)d_in[8];
  const float* initR = (const float*)d_in[10];
  const float* initT = (const float*)d_in[11];
  auto P = [&](int i) { return (const float*)d_in[i]; };
  float* out = (float*)d_out;

  // --- workspace partition (floats), ~202 MB total -------------------------
  float* wsbase = (float*)d_ws;
  size_t off = 0;
  auto alloc = [&](size_t nf) { float* p = wsbase + off; off += nf; return p; };
  float* p1t    = alloc((size_t)B_ * N_ * 3);
  float* pefeat = alloc((size_t)B_ * N_ * HIDC);
  float* pe1    = alloc((size_t)B_ * N_ * HIDC);
  float* pe2    = alloc((size_t)B_ * N_ * HIDC);
  float* x1     = alloc((size_t)B_ * NSEQ * HIDC);
  float* x2     = alloc((size_t)B_ * NSEQ * HIDC);
  WSB w;
  w.Q  = alloc((size_t)B_ * NSEQ * HIDC);
  w.K  = alloc((size_t)B_ * NSEQ * HIDC);
  w.V  = alloc((size_t)B_ * NSEQ * HIDC);
  w.AO = alloc((size_t)B_ * NSEQ * HIDC);
  w.Y  = alloc((size_t)B_ * NSEQ * HIDC);
  w.FH = alloc((size_t)B_ * NSEQ * 4 * HIDC);
  w.kv = alloc((size_t)B_ * NH * DH * DH);
  w.ks = alloc((size_t)B_ * NH * DH);
  float* s1  = alloc((size_t)B_ * SLEN * HIDC);
  float* s2  = alloc((size_t)B_ * SLEN * HIDC);
  float* s1n = alloc((size_t)B_ * SLEN * HIDC);
  float* s2n = alloc((size_t)B_ * SLEN * HIDC);
  float* f1o = alloc((size_t)B_ * NSEQ * HIDC);
  float* f2o = alloc((size_t)B_ * NSEQ * HIDC);
  float* att = alloc((size_t)B_ * NSEQ * NSEQ);
  float* p2h = alloc((size_t)B_ * N_ * 3);
  float* wb  = alloc((size_t)B_ * N_);
  float* stats = alloc((size_t)B_ * 8);
  float* Hm  = alloc((size_t)B_ * 9);
  float* rtb = alloc((size_t)B_ * 12);

  // --- positional encoding -------------------------------------------------
  k_xform<<<(B_ * N_ + 255) / 256, 256, 0, stream>>>(p1, initR, initT, p1t);
  // cloud 1: mlp1 (R=0.1, ns=32) -> ch[0:128], mlp2 (R=0.2, ns=64) -> ch[128:256]
  k_pe_mlp<<<dim3(N_ / 256, B_), 256, 0, stream>>>(p1t, P(17), P(18), P(19), P(20), P(21), P(22), 0.1f, 32, pefeat);
  k_pe_mlp<<<dim3(N_ / 256, B_), 256, 0, stream>>>(p1t, P(23), P(24), P(25), P(26), P(27), P(28), 0.2f, 64, pefeat + 128);
  gemm_flat(stream, pefeat, P(29), P(30), nullptr, pe1, B_ * N_, HIDC, HIDC, 1, 0, 1.f);  // mlp3 (W is [out,in])
  // cloud 2 (untransformed)
  k_pe_mlp<<<dim3(N_ / 256, B_), 256, 0, stream>>>(p2, P(17), P(18), P(19), P(20), P(21), P(22), 0.1f, 32, pefeat);
  k_pe_mlp<<<dim3(N_ / 256, B_), 256, 0, stream>>>(p2, P(23), P(24), P(25), P(26), P(27), P(28), 0.2f, 64, pefeat + 128);
  gemm_flat(stream, pefeat, P(29), P(30), nullptr, pe2, B_ * N_, HIDC, HIDC, 1, 0, 1.f);

  // --- input projection + bg token ----------------------------------------
  gemm(stream, f1, P(12), P(13), pe1, x1 + HIDC, N_, HIDC, HIDC,
       HIDC, HIDC, HIDC, HIDC,
       (long)N_ * HIDC, 0, 0, 0, (long)NSEQ * HIDC, 0, (long)N_ * HIDC, 0,
       B_, 1, 0, 0, 0, 1.f);
  k_bgrow<<<B_, HIDC, 0, stream>>>(x1, P(16));
  gemm(stream, f2, P(12), P(13), pe2, x2 + HIDC, N_, HIDC, HIDC,
       HIDC, HIDC, HIDC, HIDC,
       (long)N_ * HIDC, 0, 0, 0, (long)NSEQ * HIDC, 0, (long)N_ * HIDC, 0,
       B_, 1, 0, 0, 0, 1.f);
  k_bgrow<<<B_, HIDC, 0, stream>>>(x2, P(16));

  // --- sparse-to-dense transformer blocks ----------------------------------
  for (int blk = 0; blk < 3; ++blk) {
    int pbS = 31 + blk * 48, pbC = pbS + 16, pbD = pbS + 32;
    k_gather<<<dim3(SLEN, B_), HIDC, 0, stream>>>(x1, geo1, fps1, s1);
    k_gather<<<dim3(SLEN, B_), HIDC, 0, stream>>>(x2, geo2, fps2, s2);
    attn_run(stream, d_in, pbS, s1, s1, SLEN, s1, SLEN, w);
    attn_run(stream, d_in, pbS, s2, s2, SLEN, s2, SLEN, w);
    attn_run(stream, d_in, pbC, s1n, s1, SLEN, s2, SLEN, w);
    attn_run(stream, d_in, pbC, s2n, s2, SLEN, s1, SLEN, w);
    attn_run(stream, d_in, pbD, x1, x1, NSEQ, s1n, SLEN, w);
    k_copyrow0<<<B_, HIDC, 0, stream>>>(x1, s1n);
    attn_run(stream, d_in, pbD, x2, x2, NSEQ, s2n, SLEN, w);
    k_copyrow0<<<B_, HIDC, 0, stream>>>(x2, s2n);
  }

  // --- similarity, softmax, Kabsch -----------------------------------------
  gemm_flat(stream, x1, P(14), P(15), nullptr, f1o, B_ * NSEQ, HIDC, HIDC, 0, 0, 1.f);
  gemm_flat(stream, x2, P(14), P(15), nullptr, f2o, B_ * NSEQ, HIDC, HIDC, 0, 0, 1.f);
  k_rownorm<<<(B_ * NSEQ + 7) / 8, 256, 0, stream>>>(f1o, B_ * NSEQ);
  k_rownorm<<<(B_ * NSEQ + 7) / 8, 256, 0, stream>>>(f2o, B_ * NSEQ);
  gemm(stream, f1o, f2o, nullptr, nullptr, att, NSEQ, NSEQ, HIDC,
       HIDC, HIDC, NSEQ, 0,
       (long)NSEQ * HIDC, 0, (long)NSEQ * HIDC, 0, (long)NSEQ * NSEQ, 0, 0, 0,
       B_, 1, 0, 1, 0, 10.0f /* 1/TEMP */);
  k_softmax<<<dim3(N_, B_), 256, 0, stream>>>(att);
  k_p2hat<<<(B_ * N_ + 7) / 8, 256, 0, stream>>>(att, p2, p2h, wb);
  k_stats1<<<B_, 256, 0, stream>>>(wb, p1, p2h, stats);
  k_stats2<<<B_, 256, 0, stream>>>(wb, p1, p2h, stats, Hm);
  k_svd<<<1, 32, 0, stream>>>(Hm, stats, radius, rtb, out);
  k_score<<<B_, 256, 0, stream>>>(wb, p1, p2h, stats, rtb, out);
}
